// FeatureDecoder_40793599377711
// MI455X (gfx1250) — matrix operand order repaired and checked by reading, not compiled, not run on hardware
//
#include <hip/hip_runtime.h>
#include <hip/hip_bf16.h>
#include <cstdint>

#define B_    32
#define T_    512
#define IDIM_ 512
#define ODIM_ 80
#define L_    256
#define DU_   1024
#define ATT_  128
#define ACH_  10
#define AK_   31
#define PRE_  256
#define POST_ 512

typedef __bf16 bf16t;
typedef __attribute__((ext_vector_type(16))) __bf16 v16bf;
typedef __attribute__((ext_vector_type(8)))  float  v8f;

// ---------------------------------------------------------------------------
// WMMA fragment helpers (ISA 7.12.2 layouts, wave32)
// A 16x32 bf16: lane = M(0..15) + K-half(lane>>4); per chunk a lane reads two
//   4-dword runs at (rowbase + kh*4) and (+8) => 2x global_load_b128.
// B 32x16 bf16: pre-swizzled so a lane's fragment is 8 consecutive dwords
//   (dword idx = ((nt*nk + kc)*32 + lane)*8 + v) => 2x global_load_b128.
// C/D 16x16 f32: lane = N + M-half; VGPR v: m = mh*8 + v
// ---------------------------------------------------------------------------
union frag_u { v16bf bf; uint32_t u[8]; };

__device__ __forceinline__ v16bf frag_zero() {
  frag_u r;
#pragma unroll
  for (int v = 0; v < 8; ++v) r.u[v] = 0u;
  return r.bf;
}

// p = per-lane A row base (dwords) already offset by kh*4; advanced by 16/chunk
__device__ __forceinline__ v16bf load_a_ptr(const uint32_t* __restrict__ p) {
  frag_u r;
#pragma unroll
  for (int v = 0; v < 4; ++v) r.u[v] = p[v];
#pragma unroll
  for (int v = 0; v < 4; ++v) r.u[4 + v] = p[8 + v];
  return r.bf;
}

// p = per-lane swizzled B fragment base (dwords); advanced by 256/chunk
__device__ __forceinline__ v16bf load_b_ptr(const uint32_t* __restrict__ p) {
  frag_u r;
#pragma unroll
  for (int v = 0; v < 8; ++v) r.u[v] = p[v];
  return r.bf;
}

__device__ __forceinline__ float sigf(float x) { return 1.f / (1.f + expf(-x)); }

// ---------------------------------------------------------------------------
// Fused dual GEMM: C = A1(MxK1)@B1(K1xN) + A2(MxK2)@B2(K2xN) + bias
// A row-major bf16 (lda=K); B pre-swizzled dwords; C fp32 with ldc.
// One wave per 16x16 tile; 2-stage software pipeline (load k+1, wmma k).
// ---------------------------------------------------------------------------
__global__ void k_gemm_dual(const bf16t* __restrict__ A1, int K1, const uint32_t* __restrict__ B1,
                            const bf16t* __restrict__ A2, int K2, const uint32_t* __restrict__ B2,
                            const float* __restrict__ bias, float* __restrict__ C,
                            int M, int N, int ldc) {
  int lane = threadIdx.x & 31;
  int wave = (blockIdx.x * blockDim.x + threadIdx.x) >> 5;
  int Nt = N >> 4;
  int total = (M >> 4) * Nt;
  if (wave >= total) return;
  int mt = wave / Nt, nt = wave % Nt;
  int m0 = mt << 4, n0 = nt << 4;
  int lm = lane & 15, kh = lane >> 4;
  v8f acc = {0.f, 0.f, 0.f, 0.f, 0.f, 0.f, 0.f, 0.f};

  {
    int nk = K1 >> 5;
    const uint32_t* pa = (const uint32_t*)(A1 + (size_t)(m0 + lm) * K1) + (kh << 2);
    const uint32_t* pb = B1 + (((size_t)(nt * nk) * 32 + lane) << 3);
    v16bf a = load_a_ptr(pa);
    v16bf b = load_b_ptr(pb);
    for (int kc = 1; kc < nk; ++kc) {
      pa += 16; pb += 256;
      v16bf an = load_a_ptr(pa);
      v16bf bn = load_b_ptr(pb);
      acc = __builtin_amdgcn_wmma_f32_16x16x32_bf16(false, a, false, b, (short)0, acc, false, false);
      a = an; b = bn;
    }
    acc = __builtin_amdgcn_wmma_f32_16x16x32_bf16(false, a, false, b, (short)0, acc, false, false);
  }
  if (K2 > 0) {
    int nk = K2 >> 5;
    const uint32_t* pa = (const uint32_t*)(A2 + (size_t)(m0 + lm) * K2) + (kh << 2);
    const uint32_t* pb = B2 + (((size_t)(nt * nk) * 32 + lane) << 3);
    v16bf a = load_a_ptr(pa);
    v16bf b = load_b_ptr(pb);
    for (int kc = 1; kc < nk; ++kc) {
      pa += 16; pb += 256;
      v16bf an = load_a_ptr(pa);
      v16bf bn = load_b_ptr(pb);
      acc = __builtin_amdgcn_wmma_f32_16x16x32_bf16(false, a, false, b, (short)0, acc, false, false);
      a = an; b = bn;
    }
    acc = __builtin_amdgcn_wmma_f32_16x16x32_bf16(false, a, false, b, (short)0, acc, false, false);
  }

  int n  = n0 + lm;
  float bv = bias ? bias[n] : 0.f;
#pragma unroll
  for (int v = 0; v < 8; ++v) {
    int m = m0 + kh * 8 + v;
    C[(size_t)m * ldc + n] = acc[v] + bv;
  }
}

// ---------------------------------------------------------------------------
// Postnet conv1d (K=5, pad=2) as 5 shifted WMMA GEMMs over Cin, pipelined
// within each tap. Input NLC bf16 (8192 x Cin); weight swizzled per tap;
// out fp32 (8192 x Cout).
// ---------------------------------------------------------------------------
__global__ void k_conv5(const bf16t* __restrict__ in, int Cin,
                        const uint32_t* __restrict__ wt,
                        float* __restrict__ out, int Cout) {
  int lane = threadIdx.x & 31;
  int wave = (blockIdx.x * blockDim.x + threadIdx.x) >> 5;
  int Nt = Cout >> 4;
  int total = (8192 >> 4) * Nt;
  if (wave >= total) return;
  int mt = wave / Nt, nt = wave % Nt;
  int m0 = mt << 4;
  int lm = lane & 15, kh = lane >> 4;
  int m = m0 + lm, bb = m >> 8, lp = m & 255;
  int nk = Cin >> 5;
  v8f acc = {0.f, 0.f, 0.f, 0.f, 0.f, 0.f, 0.f, 0.f};
  for (int tap = 0; tap < 5; ++tap) {
    int sl = lp + tap - 2;
    bool valid = (sl >= 0 && sl < L_);
    const uint32_t* pa =
        (const uint32_t*)(in + (size_t)((bb << 8) + (valid ? sl : 0)) * Cin) + (kh << 2);
    const uint32_t* pb = wt + (size_t)tap * (Cin >> 1) * Cout + (((size_t)(nt * nk) * 32 + lane) << 3);
    v16bf a = valid ? load_a_ptr(pa) : frag_zero();
    v16bf b = load_b_ptr(pb);
    for (int kc = 1; kc < nk; ++kc) {
      pa += 16; pb += 256;
      v16bf an = valid ? load_a_ptr(pa) : frag_zero();
      v16bf bn = load_b_ptr(pb);
      acc = __builtin_amdgcn_wmma_f32_16x16x32_bf16(false, a, false, b, (short)0, acc, false, false);
      a = an; b = bn;
    }
    acc = __builtin_amdgcn_wmma_f32_16x16x32_bf16(false, a, false, b, (short)0, acc, false, false);
  }
  int n = (nt << 4) + lm;
#pragma unroll
  for (int v = 0; v < 8; ++v) {
    int mm = m0 + kh * 8 + v;
    out[(size_t)mm * Cout + n] = acc[v];
  }
}

// ---------------------------------------------------------------------------
// Weight packing: logical B(K x N) fp32 -> bf16 fragment-swizzled dwords.
// mode 0: src[k*N + n]   (already (K,N) row-major)
// mode 1: src[n*K + k]   (source is (N,K); transpose while packing)
// ---------------------------------------------------------------------------
__global__ void k_pack_b(const float* __restrict__ src, uint32_t* __restrict__ dst,
                         int K, int N, int mode) {
  int i = blockIdx.x * blockDim.x + threadIdx.x;
  if (i >= K * N) return;
  int k = i / N, n = i % N;
  float val = (mode == 0) ? src[(size_t)k * N + n] : src[(size_t)n * K + k];
  int kc = k >> 5, kk = k & 31;
  int kh = (kk >> 3) & 1, vv = ((kk & 7) >> 1) + ((kk >> 4) << 2), lo = kk & 1;
  int lane = (n & 15) + (kh << 4);
  int nt = n >> 4;
  int nk = K >> 5;
  bf16t* d = (bf16t*)dst;
  d[((((size_t)(nt * nk + kc) * 32 + lane) << 3) + vv) * 2 + lo] = (bf16t)val;
}

// conv weight (Cout, Cin, 5) fp32 -> per-tap swizzled (5 x CinP/2 x Cout dwords),
// Cin zero-padded to CinP.
__global__ void k_pack_convw(const float* __restrict__ src, uint32_t* __restrict__ dst,
                             int Cout, int Cin, int CinP) {
  int i = blockIdx.x * blockDim.x + threadIdx.x;
  int per = CinP * Cout;
  if (i >= 5 * per) return;
  int tap = i / per, rem = i % per;
  int k = rem / Cout, n = rem % Cout;
  float val = (k < Cin) ? src[((size_t)n * Cin + k) * 5 + tap] : 0.f;
  int kc = k >> 5, kk = k & 31;
  int kh = (kk >> 3) & 1, vv = ((kk & 7) >> 1) + ((kk >> 4) << 2), lo = kk & 1;
  int lane = (n & 15) + (kh << 4);
  int nt = n >> 4;
  int nk = CinP >> 5;
  bf16t* d = (bf16t*)(dst + (size_t)tap * (CinP >> 1) * Cout);
  d[((((size_t)(nt * nk + kc) * 32 + lane) << 3) + vv) * 2 + lo] = (bf16t)val;
}

// ---------------------------------------------------------------------------
// Small utility kernels
// ---------------------------------------------------------------------------
__global__ void k_cvt(const float* __restrict__ s, bf16t* __restrict__ d, int n) {
  int i = blockIdx.x * blockDim.x + threadIdx.x;
  if (i < n) d[i] = (bf16t)s[i];
}
__global__ void k_zero_f32(float* p, int n) { int i = blockIdx.x * blockDim.x + threadIdx.x; if (i < n) p[i] = 0.f; }
__global__ void k_zero_bf16(bf16t* p, int n) { int i = blockIdx.x * blockDim.x + threadIdx.x; if (i < n) p[i] = (bf16t)0.f; }

__global__ void k_aw_init(const int* __restrict__ hlens, float* __restrict__ aw) {
  int i = blockIdx.x * blockDim.x + threadIdx.x;
  if (i >= B_ * T_) return;
  int b = i / T_, t = i % T_;
  int hl = hlens[b];
  aw[i] = (t < hl) ? 1.f / (float)hl : 0.f;
}
__global__ void k_prev_init(const float* __restrict__ feat_b, float* __restrict__ prev) {
  int i = blockIdx.x * blockDim.x + threadIdx.x;
  if (i < B_ * ODIM_) prev[i] = feat_b[i % ODIM_];
}

// location-aware attention energies (per (b,t) thread)
__global__ void k_energy(const float* __restrict__ aw, const float* __restrict__ enc_proj,
                         const float* __restrict__ dec_proj, const float* __restrict__ loc_k,
                         const float* __restrict__ w_loc, const float* __restrict__ gvec,
                         const int* __restrict__ hlens, float* __restrict__ e) {
  int i = blockIdx.x * blockDim.x + threadIdx.x;
  if (i >= B_ * T_) return;
  int b = i / T_, t = i % T_;
  if (t >= hlens[b]) { e[i] = -1e30f; return; }
  float loc[ACH_];
#pragma unroll
  for (int c = 0; c < ACH_; ++c) {
    float s = 0.f;
    for (int j = 0; j < AK_; ++j) {
      int tt = t + j - (AK_ / 2);
      if (tt >= 0 && tt < T_) s += loc_k[c * AK_ + j] * aw[b * T_ + tt];
    }
    loc[c] = s;
  }
  const float* ep = enc_proj + (size_t)(b * T_ + t) * ATT_;
  const float* dp = dec_proj + b * ATT_;
  float acc = 0.f;
  for (int a = 0; a < ATT_; ++a) {
    float s = ep[a] + dp[a];
#pragma unroll
    for (int c = 0; c < ACH_; ++c) s += loc[c] * w_loc[c * ATT_ + a];
    acc += gvec[a] * tanhf(s);
  }
  e[i] = acc;
}

// per-batch softmax over T + context vector; writes bf16 context into xcat[:,0:512]
__global__ void k_softmax_attc(const float* __restrict__ e, const float* __restrict__ hs,
                               float* __restrict__ aw, bf16t* __restrict__ xcat) {
  int b = blockIdx.x, t = threadIdx.x;   // 512 threads
  __shared__ float s_w[T_];
  __shared__ float s_r[T_];
  float ev = e[b * T_ + t];
  s_r[t] = ev; __syncthreads();
  for (int s = T_ / 2; s > 0; s >>= 1) { if (t < s) s_r[t] = fmaxf(s_r[t], s_r[t + s]); __syncthreads(); }
  float mx = s_r[0]; __syncthreads();
  float w = expf(ev - mx);
  s_r[t] = w; __syncthreads();
  for (int s = T_ / 2; s > 0; s >>= 1) { if (t < s) s_r[t] += s_r[t + s]; __syncthreads(); }
  float inv = 1.f / s_r[0];
  float a = w * inv;
  aw[b * T_ + t] = a;
  s_w[t] = a; __syncthreads();
  // context: thread t handles feature dim d = t (IDIM == T_ == 512)
  float acc = 0.f;
  for (int tt = 0; tt < T_; ++tt) acc += s_w[tt] * hs[(size_t)(b * T_ + tt) * IDIM_ + t];
  xcat[b * (IDIM_ + PRE_) + t] = (bf16t)acc;
}

// prenet: relu(relu(prev@W1+b1)@W2+b2) -> xcat[:,512:768]
__global__ void k_prenet(const float* __restrict__ prev, const float* __restrict__ w1,
                         const float* __restrict__ b1, const float* __restrict__ w2,
                         const float* __restrict__ b2, bf16t* __restrict__ xcat) {
  int b = blockIdx.x, j = threadIdx.x;   // 256 threads
  __shared__ float h1[PRE_];
  float s = b1[j];
  for (int k = 0; k < ODIM_; ++k) s += prev[b * ODIM_ + k] * w1[k * PRE_ + j];
  h1[j] = fmaxf(s, 0.f);
  __syncthreads();
  float s2 = b2[j];
  for (int k = 0; k < PRE_; ++k) s2 += h1[k] * w2[k * PRE_ + j];
  xcat[b * (IDIM_ + PRE_) + IDIM_ + j] = (bf16t)fmaxf(s2, 0.f);
}

// LSTM cell 0 pointwise update
__global__ void k_cell0(const float* __restrict__ gates, float* __restrict__ c0,
                        bf16t* __restrict__ z0_bf) {
  int b = blockIdx.x, j = threadIdx.x;   // 1024 threads
  const float* g = gates + (size_t)b * 4 * DU_;
  float ii = sigf(g[j]);
  float ff = sigf(g[DU_ + j]);
  float gg = tanhf(g[2 * DU_ + j]);
  float oo = sigf(g[3 * DU_ + j]);
  int idx = b * DU_ + j;
  float c = ff * c0[idx] + ii * gg;
  c0[idx] = c;
  z0_bf[idx] = (bf16t)(oo * tanhf(c));
}

// LSTM cell 1 pointwise + stop-token prob reduction
__global__ void k_cell1_prob(const float* __restrict__ gates, float* __restrict__ c1,
                             bf16t* __restrict__ z1_bf, const float* __restrict__ prob_w,
                             const float* __restrict__ prob_b, float* __restrict__ probs, int l) {
  int b = blockIdx.x, j = threadIdx.x;   // 1024 threads
  const float* g = gates + (size_t)b * 4 * DU_;
  float ii = sigf(g[j]);
  float ff = sigf(g[DU_ + j]);
  float gg = tanhf(g[2 * DU_ + j]);
  float oo = sigf(g[3 * DU_ + j]);
  int idx = b * DU_ + j;
  float c = ff * c1[idx] + ii * gg;
  c1[idx] = c;
  float h = oo * tanhf(c);
  z1_bf[idx] = (bf16t)h;
  __shared__ float sr[DU_];
  sr[j] = h * prob_w[j];
  __syncthreads();
  for (int s = DU_ / 2; s > 0; s >>= 1) { if (j < s) sr[j] += sr[j + s]; __syncthreads(); }
  if (j == 0) probs[b * L_ + l] = sr[0] + prob_b[0];
}

// copy step output row into prev (fp32) and padded bf16 NLC buffer
__global__ void k_copyout(const float* __restrict__ outs_nlc, float* __restrict__ prev,
                          bf16t* __restrict__ outs_bf, int l) {
  int i = blockIdx.x * blockDim.x + threadIdx.x;
  if (i >= B_ * ODIM_) return;
  int b = i / ODIM_, od = i % ODIM_;
  float v = outs_nlc[(size_t)(b * L_ + l) * ODIM_ + od];
  prev[i] = v;
  outs_bf[(size_t)(b * L_ + l) * 96 + od] = (bf16t)v;
}

// training-mode BN stats over (B,L) per channel; x is (8192 x C) fp32
__global__ void k_bnstats(const float* __restrict__ x, int C, float* __restrict__ mean,
                          float* __restrict__ inv) {
  int c = blockIdx.x;
  __shared__ float s1[256];
  __shared__ float s2[256];
  float a = 0.f, q = 0.f;
  for (int r = threadIdx.x; r < 8192; r += 256) {
    float v = x[(size_t)r * C + c];
    a += v; q += v * v;
  }
  s1[threadIdx.x] = a; s2[threadIdx.x] = q; __syncthreads();
  for (int s = 128; s > 0; s >>= 1) {
    if (threadIdx.x < s) { s1[threadIdx.x] += s1[threadIdx.x + s]; s2[threadIdx.x] += s2[threadIdx.x + s]; }
    __syncthreads();
  }
  if (threadIdx.x == 0) {
    float m = s1[0] * (1.f / 8192.f);
    float v = s2[0] * (1.f / 8192.f) - m * m;
    mean[c] = m;
    inv[c] = rsqrtf(v + 1e-5f);
  }
}
__global__ void k_bnapply_tanh(const float* __restrict__ x, int C, const float* __restrict__ mean,
                               const float* __restrict__ inv, const float* __restrict__ g,
                               const float* __restrict__ bb, bf16t* __restrict__ y) {
  int i = blockIdx.x * blockDim.x + threadIdx.x;
  if (i >= 8192 * C) return;
  int c = i % C;
  y[i] = (bf16t)tanhf(g[c] * (x[i] - mean[c]) * inv[c] + bb[c]);
}

__global__ void k_final_add(const float* __restrict__ a, const float* __restrict__ b,
                            float* __restrict__ o, int n) {
  int i = blockIdx.x * blockDim.x + threadIdx.x;
  if (i < n) o[i] = a[i] + b[i];
}
__global__ void k_copy_ylens(const int* __restrict__ ylens, int* __restrict__ dst) {
  int i = blockIdx.x * blockDim.x + threadIdx.x;
  if (i < B_) dst[i] = ylens[i];
}

// ---------------------------------------------------------------------------
// Host launch
// ---------------------------------------------------------------------------
extern "C" void kernel_launch(void* const* d_in, const int* in_sizes, int n_in,
                              void* d_out, int out_size, void* d_ws, size_t ws_size,
                              hipStream_t stream) {
  const float* hs     = (const float*)d_in[0];
  const int*   hlens  = (const int*)d_in[1];
  const int*   ylens  = (const int*)d_in[3];
  const float* w_enc  = (const float*)d_in[4];
  const float* b_enc  = (const float*)d_in[5];
  const float* w_dec  = (const float*)d_in[6];
  const float* loc_k  = (const float*)d_in[7];
  const float* w_loc  = (const float*)d_in[8];
  const float* gvec   = (const float*)d_in[9];
  const float* pre_w1 = (const float*)d_in[10];
  const float* pre_b1 = (const float*)d_in[11];
  const float* pre_w2 = (const float*)d_in[12];
  const float* pre_b2 = (const float*)d_in[13];
  const float* l0_wih = (const float*)d_in[14];
  const float* l0_whh = (const float*)d_in[15];
  const float* l0_b   = (const float*)d_in[16];
  const float* l1_wih = (const float*)d_in[17];
  const float* l1_whh = (const float*)d_in[18];
  const float* l1_b   = (const float*)d_in[19];
  const float* feat_w = (const float*)d_in[20];
  const float* feat_b = (const float*)d_in[21];
  const float* prob_w = (const float*)d_in[22];
  const float* prob_b = (const float*)d_in[23];
  const float* post_k[5] = {(const float*)d_in[24], (const float*)d_in[25],
                            (const float*)d_in[26], (const float*)d_in[27],
                            (const float*)d_in[28]};
  const float* bng[4] = {(const float*)d_in[29], (const float*)d_in[31],
                         (const float*)d_in[33], (const float*)d_in[35]};
  const float* bnb[4] = {(const float*)d_in[30], (const float*)d_in[32],
                         (const float*)d_in[34], (const float*)d_in[36]};

  // workspace bump allocator
  char* ws = (char*)d_ws;
  size_t off = 0;
  auto alloc = [&](size_t bytes) { void* p = ws + off; off = (off + bytes + 255) & ~(size_t)255; return p; };

  const int XDIM = IDIM_ + PRE_;      // 768
  bf16t*    hs_bf     = (bf16t*)alloc((size_t)B_ * T_ * IDIM_ * 2);
  uint32_t* w_enc_sw  = (uint32_t*)alloc((size_t)IDIM_ * ATT_ * 2);
  uint32_t* w_dec_sw  = (uint32_t*)alloc((size_t)DU_ * ATT_ * 2);
  uint32_t* wih0_sw   = (uint32_t*)alloc((size_t)XDIM * 4 * DU_ * 2);
  uint32_t* whh0_sw   = (uint32_t*)alloc((size_t)DU_ * 4 * DU_ * 2);
  uint32_t* wih1_sw   = (uint32_t*)alloc((size_t)DU_ * 4 * DU_ * 2);
  uint32_t* whh1_sw   = (uint32_t*)alloc((size_t)DU_ * 4 * DU_ * 2);
  uint32_t* feat_w_sw = (uint32_t*)alloc((size_t)DU_ * ODIM_ * 2);
  uint32_t* w1sw      = (uint32_t*)alloc((size_t)5 * 96 * POST_ * 2);
  uint32_t* w2sw      = (uint32_t*)alloc((size_t)5 * POST_ * POST_ * 2);
  uint32_t* w3sw      = (uint32_t*)alloc((size_t)5 * POST_ * POST_ * 2);
  uint32_t* w4sw      = (uint32_t*)alloc((size_t)5 * POST_ * POST_ * 2);
  uint32_t* w5sw      = (uint32_t*)alloc((size_t)5 * POST_ * ODIM_ * 2);
  float* enc_proj  = (float*)alloc((size_t)B_ * T_ * ATT_ * 4);
  float* aw        = (float*)alloc((size_t)B_ * T_ * 4);
  float* e_buf     = (float*)alloc((size_t)B_ * T_ * 4);
  float* dec_proj  = (float*)alloc((size_t)B_ * ATT_ * 4);
  bf16t* xcat      = (bf16t*)alloc((size_t)B_ * XDIM * 2);
  float* gates     = (float*)alloc((size_t)B_ * 4 * DU_ * 4);
  float* c0        = (float*)alloc((size_t)B_ * DU_ * 4);
  float* c1        = (float*)alloc((size_t)B_ * DU_ * 4);
  bf16t* z0_bf     = (bf16t*)alloc((size_t)B_ * DU_ * 2);
  bf16t* z1_bf     = (bf16t*)alloc((size_t)B_ * DU_ * 2);
  float* prev      = (float*)alloc((size_t)B_ * ODIM_ * 4);
  float* outs_nlc  = (float*)alloc((size_t)B_ * L_ * ODIM_ * 4);
  bf16t* outs_bf   = (bf16t*)alloc((size_t)B_ * L_ * 96 * 2);
  float* conv_out  = (float*)alloc((size_t)B_ * L_ * POST_ * 4);
  bf16t* act_bf    = (bf16t*)alloc((size_t)B_ * L_ * POST_ * 2);
  float* post5     = (float*)alloc((size_t)B_ * L_ * ODIM_ * 4);
  float* bn_mean   = (float*)alloc(POST_ * 4);
  float* bn_inv    = (float*)alloc(POST_ * 4);

  auto nb = [](long n, int bs) { return (int)((n + bs - 1) / bs); };
  auto tiles = [](int M, int N) { return (M / 16) * (N / 16); };

  // ---- input conversion + weight packing (one-time per launch) ----
  k_cvt<<<nb((long)B_ * T_ * IDIM_, 256), 256, 0, stream>>>(hs, hs_bf, B_ * T_ * IDIM_);
  k_pack_b<<<nb(IDIM_ * ATT_, 256), 256, 0, stream>>>(w_enc, w_enc_sw, IDIM_, ATT_, 0);
  k_pack_b<<<nb(DU_ * ATT_, 256), 256, 0, stream>>>(w_dec, w_dec_sw, DU_, ATT_, 0);
  k_pack_b<<<nb(DU_ * ODIM_, 256), 256, 0, stream>>>(feat_w, feat_w_sw, DU_, ODIM_, 0);
  k_pack_b<<<nb(XDIM * 4 * DU_, 256), 256, 0, stream>>>(l0_wih, wih0_sw, XDIM, 4 * DU_, 1);
  k_pack_b<<<nb(DU_ * 4 * DU_, 256), 256, 0, stream>>>(l0_whh, whh0_sw, DU_, 4 * DU_, 1);
  k_pack_b<<<nb(DU_ * 4 * DU_, 256), 256, 0, stream>>>(l1_wih, wih1_sw, DU_, 4 * DU_, 1);
  k_pack_b<<<nb(DU_ * 4 * DU_, 256), 256, 0, stream>>>(l1_whh, whh1_sw, DU_, 4 * DU_, 1);
  k_pack_convw<<<nb(5 * 96 * POST_, 256), 256, 0, stream>>>(post_k[0], w1sw, POST_, ODIM_, 96);
  k_pack_convw<<<nb(5 * POST_ * POST_, 256), 256, 0, stream>>>(post_k[1], w2sw, POST_, POST_, POST_);
  k_pack_convw<<<nb(5 * POST_ * POST_, 256), 256, 0, stream>>>(post_k[2], w3sw, POST_, POST_, POST_);
  k_pack_convw<<<nb(5 * POST_ * POST_, 256), 256, 0, stream>>>(post_k[3], w4sw, POST_, POST_, POST_);
  k_pack_convw<<<nb(5 * POST_ * ODIM_, 256), 256, 0, stream>>>(post_k[4], w5sw, ODIM_, POST_, POST_);

  // ---- state init ----
  k_zero_f32<<<nb(B_ * DU_, 256), 256, 0, stream>>>(c0, B_ * DU_);
  k_zero_f32<<<nb(B_ * DU_, 256), 256, 0, stream>>>(c1, B_ * DU_);
  k_zero_bf16<<<nb(B_ * DU_, 256), 256, 0, stream>>>(z0_bf, B_ * DU_);
  k_zero_bf16<<<nb(B_ * DU_, 256), 256, 0, stream>>>(z1_bf, B_ * DU_);
  k_zero_bf16<<<nb((long)B_ * L_ * 96, 256), 256, 0, stream>>>(outs_bf, B_ * L_ * 96);
  k_aw_init<<<nb(B_ * T_, 256), 256, 0, stream>>>(hlens, aw);
  k_prev_init<<<nb(B_ * ODIM_, 256), 256, 0, stream>>>(feat_b, prev);

  // ---- encoder projection: (B*T x IDIM) @ (IDIM x ATT) + b_enc ----
  {
    int w = tiles(B_ * T_, ATT_);
    k_gemm_dual<<<nb((long)w * 32, 256), 256, 0, stream>>>(
        hs_bf, IDIM_, w_enc_sw, nullptr, 0, nullptr, b_enc, enc_proj, B_ * T_, ATT_, ATT_);
  }

  float* probs_out = (float*)d_out + (size_t)B_ * L_ * ODIM_;

  // ---- decoder scan (256 sequential steps) ----
  for (int l = 0; l < L_; ++l) {
    // dec_proj = z0 @ w_dec
    k_gemm_dual<<<nb((long)tiles(B_, ATT_) * 32, 256), 256, 0, stream>>>(
        z0_bf, DU_, w_dec_sw, nullptr, 0, nullptr, nullptr, dec_proj, B_, ATT_, ATT_);
    // location-aware attention energies + softmax + context
    k_energy<<<nb(B_ * T_, 128), 128, 0, stream>>>(aw, enc_proj, dec_proj, loc_k, w_loc,
                                                   gvec, hlens, e_buf);
    k_softmax_attc<<<B_, T_, 0, stream>>>(e_buf, hs, aw, xcat);
    // prenet on previous prediction
    k_prenet<<<B_, PRE_, 0, stream>>>(prev, pre_w1, pre_b1, pre_w2, pre_b2, xcat);
    // LSTM0 gates = xcat@Wih0^T + z0@Whh0^T + b0   (fused dual WMMA GEMM)
    k_gemm_dual<<<nb((long)tiles(B_, 4 * DU_) * 32, 256), 256, 0, stream>>>(
        xcat, XDIM, wih0_sw, z0_bf, DU_, whh0_sw, l0_b, gates, B_, 4 * DU_, 4 * DU_);
    k_cell0<<<B_, DU_, 0, stream>>>(gates, c0, z0_bf);
    // LSTM1 gates = z0@Wih1^T + z1@Whh1^T + b1
    k_gemm_dual<<<nb((long)tiles(B_, 4 * DU_) * 32, 256), 256, 0, stream>>>(
        z0_bf, DU_, wih1_sw, z1_bf, DU_, whh1_sw, l1_b, gates, B_, 4 * DU_, 4 * DU_);
    k_cell1_prob<<<B_, DU_, 0, stream>>>(gates, c1, z1_bf, prob_w, prob_b, probs_out, l);
    // feat projection: out = z1 @ feat_w + feat_b, stored at NLC row (b*L + l)
    k_gemm_dual<<<nb((long)tiles(B_, ODIM_) * 32, 256), 256, 0, stream>>>(
        z1_bf, DU_, feat_w_sw, nullptr, 0, nullptr, feat_b, outs_nlc + (size_t)l * ODIM_,
        B_, ODIM_, L_ * ODIM_);
    k_copyout<<<nb(B_ * ODIM_, 256), 256, 0, stream>>>(outs_nlc, prev, outs_bf, l);
  }

  // ---- postnet: 4x (conv5 -> BN(train) -> tanh), then conv5, residual add ----
  const uint32_t* wmid[3] = {w2sw, w3sw, w4sw};
  {
    int w = tiles(8192, POST_);
    k_conv5<<<nb((long)w * 32, 256), 256, 0, stream>>>(outs_bf, 96, w1sw, conv_out, POST_);
    k_bnstats<<<POST_, 256, 0, stream>>>(conv_out, POST_, bn_mean, bn_inv);
    k_bnapply_tanh<<<nb((long)8192 * POST_, 256), 256, 0, stream>>>(
        conv_out, POST_, bn_mean, bn_inv, bng[0], bnb[0], act_bf);
    for (int layer = 1; layer < 4; ++layer) {
      k_conv5<<<nb((long)w * 32, 256), 256, 0, stream>>>(act_bf, POST_, wmid[layer - 1],
                                                         conv_out, POST_);
      k_bnstats<<<POST_, 256, 0, stream>>>(conv_out, POST_, bn_mean, bn_inv);
      k_bnapply_tanh<<<nb((long)8192 * POST_, 256), 256, 0, stream>>>(
          conv_out, POST_, bn_mean, bn_inv, bng[layer], bnb[layer], act_bf);
    }
    int w5 = tiles(8192, ODIM_);
    k_conv5<<<nb((long)w5 * 32, 256), 256, 0, stream>>>(act_bf, POST_, w5sw, post5, ODIM_);
  }

  // ---- outputs: outs = dec_out + postnet residual (already NLC = (B,L,ODIM)) ----
  k_final_add<<<nb((long)B_ * L_ * ODIM_, 256), 256, 0, stream>>>(
      outs_nlc, post5, (float*)d_out, B_ * L_ * ODIM_);
  k_copy_ylens<<<1, 32, 0, stream>>>(
      ylens, (int*)((float*)d_out + (size_t)B_ * L_ * ODIM_ + (size_t)B_ * L_));
}
